// AttentionConv_32487132627486
// MI455X (gfx1250) — matrix operand order repaired and checked by reading, not compiled, not run on hardware
//
#include <hip/hip_runtime.h>
#include <hip/hip_bf16.h>
#include <math.h>

typedef __attribute__((ext_vector_type(16))) __bf16 v16bf;
typedef __attribute__((ext_vector_type(8)))  float  v8f;

#define B_    32
#define CIN_  64
#define H_    32
#define W_    32
#define CO_   64
#define G_    8
#define CPG_  8
#define PAD_  3
#define HP_   38
#define WP_   38
#define NPOS_ (HP_*WP_)   /* 1444 padded positions */
#define HWQ_  (H_*W_)     /* 1024 */
#define PTILE_ 64         /* positions per block (4 WMMA subtiles of 16) */
#define XSROW_ (CIN_ + 8) /* LDS row stride (bf16): 144B, 16B-aligned */

// ---------------------------------------------------------------------------
// Kernel 1: Q/K/V 1x1-conv projections as WMMA bf16 GEMMs.
//   Per workgroup: batch b, one 64-position tile of the padded 38x38 domain.
//   4 waves; wave w owns output channels [16w, 16w+16).
//   Weight A-fragments (Q,K,V x two K-chunks) are built ONCE per wave and
//   reused across 4 position subtiles -> 24 WMMAs per wave per block.
//   LDS tile is POSITION-MAJOR (xs[pos][cin]) so each lane's B-fragment is a
//   contiguous, aligned 32B run -> two ds_load_b128 per fragment.
//   Outputs are stored CHANNEL-LAST ([pos][co]) so the attention kernel can
//   read each group's 8 channels with two b128 loads, and so the D-fragment
//   epilogue (8 consecutive channels per lane) becomes two float4 stores.
// ---------------------------------------------------------------------------
__global__ __launch_bounds__(128) void proj_wmma_kernel(
    const float* __restrict__ x,
    const float* __restrict__ wq,
    const float* __restrict__ wk,
    const float* __restrict__ wv,
    float* __restrict__ Qo,   // (B, 32*32, 64)   pos-major, channel-last
    float* __restrict__ Kp,   // (B, 38*38, 64)
    float* __restrict__ Vp)   // (B, 38*38, 64)
{
    __shared__ __attribute__((aligned(16))) __bf16 xs[PTILE_][XSROW_];

    const int b   = blockIdx.y;
    const int p0  = blockIdx.x * PTILE_;
    const int tid = threadIdx.x;

    // Stage zero-padded activation tile transposed to [pos][cin] as bf16.
    // Each thread converts a channel PAIR -> one packed ds_store_b32.
    // Threads sweep positions fastest -> global reads stay coalesced.
    for (int e = tid; e < (CIN_ / 2) * PTILE_; e += 128) {
        int cpair = e >> 6;          // 0..31 -> channels {2cpair, 2cpair+1}
        int pos   = e & 63;
        int p     = p0 + pos;
        float v0 = 0.f, v1 = 0.f;
        if (p < NPOS_) {
            int hp = p / WP_, wp = p - hp * WP_;
            int h  = hp - PAD_, w = wp - PAD_;
            if ((unsigned)h < (unsigned)H_ && (unsigned)w < (unsigned)W_) {
                const float* src = x + ((size_t)(b * CIN_ + 2 * cpair) * H_ + h) * W_ + w;
                v0 = src[0];
                v1 = src[HWQ_];      // next channel, same position
            }
        }
        xs[pos][2 * cpair]     = (__bf16)v0;
        xs[pos][2 * cpair + 1] = (__bf16)v1;
    }
    __syncthreads();

    const int lane = tid & 31;
    const int co0  = (tid >> 5) * 16;   // wave id -> CO tile
    const int half = lane >> 4;         // 0: lanes 0-15, 1: lanes 16-31
    const int mn   = lane & 15;         // A: row M;  B: col N;  D: col N

    // Build all six A fragments once (Q,K,V x K-chunks); reused 4x below.
    // A fragment (16x32 bf16, M x K): lane row M = mn;
    // K = kc*32 + 16*half + e   (contiguous, matches B fragment)
    const float* wmat[3] = { wq, wk, wv };
    v16bf afrag[3][2];
    #pragma unroll
    for (int m = 0; m < 3; ++m) {
        #pragma unroll
        for (int kc = 0; kc < 2; ++kc) {
            #pragma unroll
            for (int e = 0; e < 16; ++e) {
                int k = kc * 32 + half * 16 + e;
                afrag[m][kc][e] = (__bf16)wmat[m][(co0 + mn) * CIN_ + k];
            }
        }
    }

    // 4 position subtiles of 16.
    #pragma unroll
    for (int st = 0; st < 4; ++st) {
        // B fragment (32x16 bf16, K x N): lane column N = mn, element e holds
        // K = kc*32 + 16*half + e  -> 16 CONTIGUOUS bf16 (32B, 16B-aligned).
        int bpos = st * 16 + mn;
        v16bf bfrag0 = *(const v16bf*)(&xs[bpos][half * 16]);
        v16bf bfrag1 = *(const v16bf*)(&xs[bpos][32 + half * 16]);

        v8f acc[3];
        #pragma unroll
        for (int m = 0; m < 3; ++m) {
            v8f c = {0.f, 0.f, 0.f, 0.f, 0.f, 0.f, 0.f, 0.f};
            c = __builtin_amdgcn_wmma_f32_16x16x32_bf16(
                    false, afrag[m][0], false, bfrag0, (short)0, c, false, false);
            c = __builtin_amdgcn_wmma_f32_16x16x32_bf16(
                    false, afrag[m][1], false, bfrag1, (short)0, c, false, false);
            acc[m] = c;
        }

        // Store: D elem r -> (M = r + 8*half, N = mn). The 8 elements of one
        // lane are 8 CONSECUTIVE output channels -> two float4 stores each.
        int pos = p0 + st * 16 + mn;
        if (pos < NPOS_) {
            int cbase = co0 + half * 8;    // multiple of 8 -> 32B aligned
            float4 k_lo = make_float4(acc[1][0], acc[1][1], acc[1][2], acc[1][3]);
            float4 k_hi = make_float4(acc[1][4], acc[1][5], acc[1][6], acc[1][7]);
            float4 v_lo = make_float4(acc[2][0], acc[2][1], acc[2][2], acc[2][3]);
            float4 v_hi = make_float4(acc[2][4], acc[2][5], acc[2][6], acc[2][7]);
            float* kdst = Kp + ((size_t)(b * NPOS_ + pos)) * CO_ + cbase;
            float* vdst = Vp + ((size_t)(b * NPOS_ + pos)) * CO_ + cbase;
            *(float4*)(kdst)     = k_lo;
            *(float4*)(kdst + 4) = k_hi;
            *(float4*)(vdst)     = v_lo;
            *(float4*)(vdst + 4) = v_hi;

            int hp = pos / WP_, wp = pos - hp * WP_;
            int h  = hp - PAD_, w = wp - PAD_;
            if (((unsigned)h < (unsigned)H_) && ((unsigned)w < (unsigned)W_)) {
                float4 q_lo = make_float4(acc[0][0], acc[0][1], acc[0][2], acc[0][3]);
                float4 q_hi = make_float4(acc[0][4], acc[0][5], acc[0][6], acc[0][7]);
                float* qdst = Qo + ((size_t)(b * HWQ_ + h * W_ + w)) * CO_ + cbase;
                *(float4*)(qdst)     = q_lo;
                *(float4*)(qdst + 4) = q_hi;
            }
        }
    }
}

// ---------------------------------------------------------------------------
// Kernel 2: 7x7 sliding-window grouped attention + analytic ring mask.
//   One thread per (b, g, h, w). 49 logits in registers.
//   K/V/Q are channel-last: each tap is two b128 loads.
//   rel bias is separable: sum_c q_c*(k_c+rel_c) = q.k + bias[i or j],
//   with bias precomputed per thread (g<4 -> rel_h(i), g>=4 -> rel_w(j)).
// ---------------------------------------------------------------------------
__global__ __launch_bounds__(256) void attn_window_kernel(
    const float* __restrict__ Qo,      // (B,1024,64)
    const float* __restrict__ Kp,      // (B,1444,64)
    const float* __restrict__ Vp,      // (B,1444,64)
    const float* __restrict__ rel_h,   // (32,7)
    const float* __restrict__ rel_w,   // (32,7)
    const float* __restrict__ cv,      // (8,)
    float* __restrict__ out)           // (B,64,32,32)
{
    int idx = blockIdx.x * 256 + threadIdx.x;
    int w = idx & 31;
    int h = (idx >> 5) & 31;
    int g = (idx >> 10) & 7;   // uniform within a wave
    int b = idx >> 13;

    const float4* qv =
        (const float4*)(Qo + ((size_t)(b * HWQ_ + h * W_ + w)) * CO_ + g * CPG_);
    float4 q0 = qv[0], q1 = qv[1];
    float q[CPG_] = { q0.x, q0.y, q0.z, q0.w, q1.x, q1.y, q1.z, q1.w };

    // Separable rel bias: bias[t] = sum_c q[c] * rel[c,t]
    const float* rel = (g < 4) ? (rel_h + g * CPG_ * 7)
                               : (rel_w + (g - 4) * CPG_ * 7);
    float bias[7];
    #pragma unroll
    for (int t = 0; t < 7; ++t) {
        float s = 0.f;
        #pragma unroll
        for (int c = 0; c < CPG_; ++c)
            s += q[c] * rel[c * 7 + t];
        bias[t] = s;
    }

    // Pass 1: 49 logits, track max.
    float sc[49];
    float smax = -1e30f;
    #pragma unroll
    for (int i = 0; i < 7; ++i) {
        #pragma unroll
        for (int j = 0; j < 7; ++j) {
            int pp = (h + i) * WP_ + (w + j);
            const float4* kv =
                (const float4*)(Kp + ((size_t)(b * NPOS_ + pp)) * CO_ + g * CPG_);
            float4 ka = kv[0], kb = kv[1];
            float dot = q[0] * ka.x + q[1] * ka.y + q[2] * ka.z + q[3] * ka.w
                      + q[4] * kb.x + q[5] * kb.y + q[6] * kb.z + q[7] * kb.w
                      + ((g < 4) ? bias[i] : bias[j]);
            sc[i * 7 + j] = dot;
            smax = fmaxf(smax, dot);
        }
    }

    // Softmax over the 49 taps.
    float ssum = 0.f;
    #pragma unroll
    for (int t = 0; t < 49; ++t) {
        sc[t] = expf(sc[t] - smax);
        ssum += sc[t];
    }
    float inv = 1.f / ssum;

    // Pass 2: attn-weighted V accumulation.
    float accv[CPG_] = {};
    #pragma unroll
    for (int i = 0; i < 7; ++i) {
        #pragma unroll
        for (int j = 0; j < 7; ++j) {
            float a  = sc[i * 7 + j] * inv;
            int   pp = (h + i) * WP_ + (w + j);
            const float4* vv =
                (const float4*)(Vp + ((size_t)(b * NPOS_ + pp)) * CO_ + g * CPG_);
            float4 va = vv[0], vb = vv[1];
            accv[0] += a * va.x;  accv[1] += a * va.y;
            accv[2] += a * va.z;  accv[3] += a * va.w;
            accv[4] += a * vb.x;  accv[5] += a * vb.y;
            accv[6] += a * vb.z;  accv[7] += a * vb.w;
        }
    }

    // Adaptive ring mask (closed form of _adaptive_mask).
    int  r    = min(h, 31 - h);
    int  lo   = (h <= 31 - h) ? r : r + 1;
    int  hi   = 31 - r;
    bool ring = (w >= lo) && (w <= hi);
    float om  = ((float)(r - 15) + cv[g] * 16.f) * (1.f / 3.f) + 1.f;
    om = fminf(fmaxf(om, 0.f), 1.f);
    float mval = ring ? om : 1.f;

    #pragma unroll
    for (int c = 0; c < CPG_; ++c)
        out[(b * CO_ + g * CPG_ + c) * HWQ_ + h * W_ + w] = accv[c] * mval;
}

// ---------------------------------------------------------------------------
extern "C" void kernel_launch(void* const* d_in, const int* in_sizes, int n_in,
                              void* d_out, int out_size, void* d_ws, size_t ws_size,
                              hipStream_t stream)
{
    const float* x  = (const float*)d_in[0];
    const float* wq = (const float*)d_in[1];
    const float* wk = (const float*)d_in[2];
    const float* wv = (const float*)d_in[3];
    const float* rh = (const float*)d_in[4];
    const float* rw = (const float*)d_in[5];
    const float* cv = (const float*)d_in[6];

    // Workspace: Q (8 MB) + Kpad (11.8 MB) + Vpad (11.8 MB), all L2-resident.
    float* Qo = (float*)d_ws;
    float* Kp = Qo + (size_t)B_ * HWQ_ * CO_;
    float* Vp = Kp + (size_t)B_ * NPOS_ * CO_;

    dim3 grid1((NPOS_ + PTILE_ - 1) / PTILE_, B_);
    proj_wmma_kernel<<<grid1, 128, 0, stream>>>(x, wq, wk, wv, Qo, Kp, Vp);

    attn_window_kernel<<<(B_ * G_ * H_ * W_) / 256, 256, 0, stream>>>(
        Qo, Kp, Vp, rh, rw, cv, (float*)d_out);
}
